// RK4Simulator_6794638262569
// MI455X (gfx1250) — matrix operand order repaired and checked by reading, not compiled, not run on hardware
//
#include <hip/hip_runtime.h>
#include <cstdint>

// ---------------- problem constants ----------------
constexpr int kNX = 64, kNU = 32, kNY = 16, kNH = 128;
constexpr int kB = 512, kT = 512;
constexpr float kTS = 0.01f;

// ---------------- WMMA types ----------------
typedef __attribute__((ext_vector_type(16))) _Float16 v16h;
typedef __attribute__((ext_vector_type(8)))  float    v8f;
typedef __attribute__((ext_vector_type(4)))  float    v4f;
typedef __attribute__((ext_vector_type(2)))  _Float16 h2;

// Fragment table (A-operands, all weights pre-transposed so batch is the N dim)
constexpr int FRAG_W1X = 0;   // 8 mt x 2 kc   (Hid x State)
constexpr int FRAG_W1U = 16;  // 8 mt          (Hid x Input)
constexpr int FRAG_W2  = 24;  // 4 mt x 4 kc   (State x Hid)
constexpr int FRAG_C   = 40;  // 2 kc          (Ny x State)
constexpr int FRAG_D   = 42;  // 1             (Ny x Input)
constexpr int NFRAG    = 43;
constexpr int FRAG_HALVES = 512;

__device__ __forceinline__ float fast_tanh(float x) {
#if __has_builtin(__builtin_amdgcn_tanhf)
  return __builtin_amdgcn_tanhf(x);   // gfx1250 v_tanh_f32
#else
  return tanhf(x);
#endif
}

__device__ __forceinline__ v8f wmma_f16(v16h a, v16h b, v8f c) {
  return __builtin_amdgcn_wmma_f32_16x16x32_f16(false, a, false, b, (short)0, c,
                                                false, false);
}

__device__ __forceinline__ v16h load_frag(const _Float16* sF, int f, int lane) {
  return *(const v16h*)(sF + f * FRAG_HALVES + lane * 16);
}

// two f32 -> one dword of packed f16 (lowers to v_cvt_pk_f16_f32)
__device__ __forceinline__ unsigned pack2h(float x, float y) {
  union { h2 h; unsigned u; } c;
  c.h[0] = (_Float16)x;
  c.h[1] = (_Float16)y;
  return c.u;
}

// Half-wave swap of a 32-bit register pair: on output
//   a' = [a(lanes 0-15) | b(lanes 0-15)]   b' = [a(lanes 16-31) | b(lanes 16-31)]
// One v_permlane16_swap_b32 on gfx1250.
__device__ __forceinline__ void half_swap_u32(unsigned& a, unsigned& b, bool lo) {
#if __has_builtin(__builtin_amdgcn_permlane16_swap)
  typedef unsigned uv2 __attribute__((ext_vector_type(2)));
  uv2 r = __builtin_amdgcn_permlane16_swap(a, b, false, false);
  a = r.x; b = r.y;
#elif __has_builtin(__builtin_amdgcn_permlanex16)
  unsigned pa = __builtin_amdgcn_permlanex16(a, a, 0x76543210u, 0xFEDCBA98u,
                                             false, false);
  unsigned pb = __builtin_amdgcn_permlanex16(b, b, 0x76543210u, 0xFEDCBA98u,
                                             false, false);
  unsigned na = lo ? a : pb;
  unsigned nb = lo ? pa : b;
  a = na; b = nb;
#else
  unsigned pa = (unsigned)__shfl_xor((int)a, 16, 32);
  unsigned pb = (unsigned)__shfl_xor((int)b, 16, 32);
  unsigned na = lo ? a : pb;
  unsigned nb = lo ? pa : b;
  a = na; b = nb;
#endif
}

// Build a B-operand fragment (K=32 chunk) from two 16-row accumulator tiles.
// Accumulator: lane L holds N=L&15, M=(L>>4)*8+r.  B: lane L holds N=L&15,
// K=(L>>4)*16+e.  Convert to packed f16 first, then swap dwords: 8 cvt + 4 swap.
__device__ __forceinline__ v16h pack_B(const float tE[8], const float tO[8],
                                       bool lo) {
  union { v16h h; unsigned u[8]; } f;
#pragma unroll
  for (int r = 0; r < 4; ++r) {
    unsigned a = pack2h(tE[2 * r], tE[2 * r + 1]);
    unsigned b = pack2h(tO[2 * r], tO[2 * r + 1]);
    half_swap_u32(a, b, lo);
    f.u[r]     = a;   // elements 2r, 2r+1      (K-local 0..7 group)
    f.u[r + 4] = b;   // elements 8+2r, 9+2r    (K-local 8..15 group)
  }
  return f.h;
}

// B-operand view of (x + coef*k), accumulator layout in, fragment out.
__device__ __forceinline__ void build_xB(const float x[4][8], const float k[4][8],
                                         float coef, bool lo, v16h xB[2]) {
#pragma unroll
  for (int c = 0; c < 2; ++c) {
    float tE[8], tO[8];
#pragma unroll
    for (int r = 0; r < 8; ++r) {
      tE[r] = __builtin_fmaf(coef, k[2 * c][r],     x[2 * c][r]);
      tO[r] = __builtin_fmaf(coef, k[2 * c + 1][r], x[2 * c + 1][r]);
    }
    xB[c] = pack_B(tE, tO, lo);
  }
}

// B-operand view of x itself (coef == 0 path, no FMAs).
__device__ __forceinline__ void build_xB0(const float x[4][8], bool lo,
                                          v16h xB[2]) {
#pragma unroll
  for (int c = 0; c < 2; ++c)
    xB[c] = pack_B(x[2 * c], x[2 * c + 1], lo);
}

// dx = W2^T tanh(W1x^T x + W1u^T u + b1) + b2 for a 16-batch tile (batch = N).
// Biases enter as the WMMA accumulator initial value (zero extra VALU).
__device__ __forceinline__ void ss_dx(const _Float16* sF, const v8f b1a[8],
                                      const v8f b2a[4], const v16h xB[2],
                                      v16h uB, int lane, bool lo,
                                      float kout[4][8]) {
  v16h hB[4];
#pragma unroll
  for (int c = 0; c < 4; ++c) {           // hidden dim in 32-wide chunks
    float tile[2][8];
#pragma unroll
    for (int p = 0; p < 2; ++p) {
      int mt = 2 * c + p;                 // 16-row hidden tile
      v8f h = b1a[mt];                    // C := b1 broadcast tile
      h = wmma_f16(load_frag(sF, FRAG_W1X + mt * 2 + 0, lane), xB[0], h);
      h = wmma_f16(load_frag(sF, FRAG_W1X + mt * 2 + 1, lane), xB[1], h);
      h = wmma_f16(load_frag(sF, FRAG_W1U + mt, lane), uB, h);
#pragma unroll
      for (int r = 0; r < 8; ++r) tile[p][r] = fast_tanh(h[r]);
    }
    hB[c] = pack_B(tile[0], tile[1], lo);
  }
#pragma unroll
  for (int mt = 0; mt < 4; ++mt) {        // state tiles
    v8f k = b2a[mt];                      // C := b2 broadcast tile
#pragma unroll
    for (int c = 0; c < 4; ++c)
      k = wmma_f16(load_frag(sF, FRAG_W2 + mt * 4 + c, lane), hB[c], k);
#pragma unroll
    for (int r = 0; r < 8; ++r) kout[mt][r] = k[r];
  }
}

// ---------------- weight packing (runs every launch, deterministic) --------
__global__ __launch_bounds__(512) void prep_kernel(
    const float* __restrict__ W1x, const float* __restrict__ W1u,
    const float* __restrict__ W2,  const float* __restrict__ Cm,
    const float* __restrict__ Dm,  const float* __restrict__ b1,
    const float* __restrict__ b2,  _Float16* __restrict__ frag,
    float* __restrict__ oB1, float* __restrict__ oB2) {
  int f = blockIdx.x;
  int tid = threadIdx.x;
  if (f == NFRAG) {                       // bias pass-through copies
    if (tid < kNH) oB1[tid] = b1[tid];
    else if (tid < kNH + kNX) oB2[tid - kNH] = b2[tid - kNH];
    return;
  }
  int L = tid >> 4, i = tid & 15;
  int m  = L & 15;                        // tile row (hidden/state/ny index)
  int kg = L >> 4;
  int v = i >> 1, s = i & 1;
  // 16-bit A-operand layout: lane L, element i -> local K index
  int kl = kg * 16 + i;
  float val;
  if (f < 16) {                           // W1x^T : A[m=hid][k=state]
    int mt = f >> 1, kc = f & 1;
    val = W1x[(kc * 32 + kl) * kNH + (mt * 16 + m)];
  } else if (f < 24) {                    // W1u^T : A[m=hid][k=input]
    int mt = f - 16;
    val = W1u[kl * kNH + (mt * 16 + m)];
  } else if (f < 40) {                    // W2^T : A[m=state][k=hid]
    int g = f - 24, mt = g >> 2, kc = g & 3;
    val = W2[(kc * 32 + kl) * kNX + (mt * 16 + m)];
  } else if (f < 42) {                    // C^T : A[m=ny][k=state]
    int kc = f - 40;
    val = Cm[(kc * 32 + kl) * kNY + m];
  } else {                                // D^T : A[m=ny][k=input]
    val = Dm[kl * kNY + m];
  }
  frag[f * FRAG_HALVES + L * 16 + i] = (_Float16)val;
}

// ---------------- main RK4 scan: 1 wave per 16-batch tile ------------------
__global__ __launch_bounds__(32) void rk4_kernel(
    const float* __restrict__ u_batch, const float* __restrict__ x0,
    const _Float16* __restrict__ wsFrag, const float* __restrict__ wsB1,
    const float* __restrict__ wsB2, float* __restrict__ Xout,
    float* __restrict__ Yout) {
  __shared__ __align__(32) _Float16 sF[NFRAG * FRAG_HALVES];

  const int lane = threadIdx.x;
  // stage packed weights into LDS once (43 KB)
  {
    const uint4* src = (const uint4*)wsFrag;
    uint4* dst = (uint4*)sF;
    const int nvec = NFRAG * FRAG_HALVES * 2 / 16;
    for (int i = lane; i < nvec; i += 32) dst[i] = src[i];
  }
  __syncthreads();

  const int n   = lane & 15;
  const int kg  = lane >> 4;
  const int kg8 = kg * 8;
  const bool lo = lane < 16;
  const int b   = blockIdx.x * 16 + n;    // this lane's batch column

  const float dt2 = kTS * 0.5f, dt6 = kTS / 6.0f;

  // bias tiles pre-broadcast into accumulator layout (loop-invariant)
  v8f b1a[8], b2a[4];
#pragma unroll
  for (int mt = 0; mt < 8; ++mt) {
    const v4f* p = (const v4f*)(wsB1 + mt * 16 + kg8);
    v4f a0 = p[0], a1 = p[1];
#pragma unroll
    for (int r = 0; r < 4; ++r) { b1a[mt][r] = a0[r]; b1a[mt][r + 4] = a1[r]; }
  }
#pragma unroll
  for (int mt = 0; mt < 4; ++mt) {
    const v4f* p = (const v4f*)(wsB2 + mt * 16 + kg8);
    v4f a0 = p[0], a1 = p[1];
#pragma unroll
    for (int r = 0; r < 4; ++r) { b2a[mt][r] = a0[r]; b2a[mt][r + 4] = a1[r]; }
  }

  // state x in accumulator layout: xacc[mt][r] = x[b][mt*16 + kg*8 + r]
  float xacc[4][8];
#pragma unroll
  for (int mt = 0; mt < 4; ++mt) {
    const v4f* p = (const v4f*)(x0 + (size_t)b * kNX + mt * 16 + kg8);
    v4f a0 = p[0], a1 = p[1];
#pragma unroll
    for (int r = 0; r < 4; ++r) { xacc[mt][r] = a0[r]; xacc[mt][r + 4] = a1[r]; }
  }

  // u as B-fragment: lane L reads 16 consecutive input features (64B, aligned)
  const float* ubase = u_batch + (size_t)b * kT * kNU + kg * 16;
  v16h uB;
  {
    const v4f* p = (const v4f*)ubase;
    v16h f;
#pragma unroll
    for (int q = 0; q < 4; ++q) {
      v4f w = p[q];
#pragma unroll
      for (int r = 0; r < 4; ++r) f[q * 4 + r] = (_Float16)w[r];
    }
    uB = f;
  }

  float kA[4][8], kBuf[4][8], kS[4][8];
  v16h xB[2];

  for (int t = 0; t < kT; ++t) {
    // software prefetch of next step's inputs over this step's compute
    v4f unext[4];
    if (t + 1 < kT) {
      const v4f* p = (const v4f*)(ubase + (size_t)(t + 1) * kNU);
#pragma unroll
      for (int q = 0; q < 4; ++q) unext[q] = p[q];
    }

    build_xB0(xacc, lo, xB);              // B-operand view of current x

    // X_sim stores the pre-update state (2x 16B contiguous per lane per tile)
#pragma unroll
    for (int mt = 0; mt < 4; ++mt) {
      v4f* p = (v4f*)(Xout + ((size_t)b * kT + t) * kNX + mt * 16 + kg8);
      v4f a0, a1;
#pragma unroll
      for (int r = 0; r < 4; ++r) { a0[r] = xacc[mt][r]; a1[r] = xacc[mt][r + 4]; }
      p[0] = a0; p[1] = a1;
    }
    // y = C^T x + D^T u
    {
      v8f y = {};
      y = wmma_f16(load_frag(sF, FRAG_C + 0, lane), xB[0], y);
      y = wmma_f16(load_frag(sF, FRAG_C + 1, lane), xB[1], y);
      y = wmma_f16(load_frag(sF, FRAG_D, lane), uB, y);
      v4f* p = (v4f*)(Yout + ((size_t)b * kT + t) * kNY + kg8);
      v4f y0, y1;
#pragma unroll
      for (int r = 0; r < 4; ++r) { y0[r] = y[r]; y1[r] = y[r + 4]; }
      p[0] = y0; p[1] = y1;
    }

    // RK4 (k1 lands directly in the kS accumulator)
    ss_dx(sF, b1a, b2a, xB, uB, lane, lo, kS);                   // k1

    build_xB(xacc, kS, dt2, lo, xB);
    ss_dx(sF, b1a, b2a, xB, uB, lane, lo, kBuf);                 // k2
#pragma unroll
    for (int mt = 0; mt < 4; ++mt)
#pragma unroll
      for (int r = 0; r < 8; ++r) kS[mt][r] += 2.0f * kBuf[mt][r];

    build_xB(xacc, kBuf, dt2, lo, xB);
    ss_dx(sF, b1a, b2a, xB, uB, lane, lo, kA);                   // k3
#pragma unroll
    for (int mt = 0; mt < 4; ++mt)
#pragma unroll
      for (int r = 0; r < 8; ++r) kS[mt][r] += 2.0f * kA[mt][r];

    build_xB(xacc, kA, kTS, lo, xB);
    ss_dx(sF, b1a, b2a, xB, uB, lane, lo, kBuf);                 // k4
#pragma unroll
    for (int mt = 0; mt < 4; ++mt)
#pragma unroll
      for (int r = 0; r < 8; ++r) {
        kS[mt][r] += kBuf[mt][r];
        xacc[mt][r] += dt6 * kS[mt][r];
      }

    if (t + 1 < kT) {
      v16h f;
#pragma unroll
      for (int q = 0; q < 4; ++q)
#pragma unroll
        for (int r = 0; r < 4; ++r) f[q * 4 + r] = (_Float16)unext[q][r];
      uB = f;
    }
  }
}

// ---------------- host entry ----------------
extern "C" void kernel_launch(void* const* d_in, const int* in_sizes, int n_in,
                              void* d_out, int out_size, void* d_ws,
                              size_t ws_size, hipStream_t stream) {
  (void)in_sizes; (void)n_in; (void)out_size; (void)ws_size;
  const float* u   = (const float*)d_in[0];
  const float* x0  = (const float*)d_in[1];
  const float* W1x = (const float*)d_in[2];
  const float* W1u = (const float*)d_in[3];
  const float* b1  = (const float*)d_in[4];
  const float* W2  = (const float*)d_in[5];
  const float* b2  = (const float*)d_in[6];
  const float* Cm  = (const float*)d_in[7];
  const float* Dm  = (const float*)d_in[8];

  _Float16* frag = (_Float16*)d_ws;
  float* oB1 = (float*)((char*)d_ws + NFRAG * FRAG_HALVES * 2);
  float* oB2 = oB1 + kNH;

  float* Xout = (float*)d_out;
  float* Yout = Xout + (size_t)kB * kT * kNX;

  prep_kernel<<<NFRAG + 1, 512, 0, stream>>>(W1x, W1u, W2, Cm, Dm, b1, b2,
                                             frag, oB1, oB2);
  rk4_kernel<<<kB / 16, 32, 0, stream>>>(u, x0, frag, oB1, oB2, Xout, Yout);
}
